// HyperDomainFusionModule_6493990551559
// MI455X (gfx1250) — matrix operand order repaired and checked by reading, not compiled, not run on hardware
//
#include <hip/hip_runtime.h>
#include <hip/hip_bf16.h>

// ---------------------------------------------------------------------------
// HyperDomainFusionModule for MI455X (gfx1250, wave32, WMMA + async LDS).
//
// bank is dead code in the reference (top-2 indices only address rows <4096).
// All GEMMs: bf16 WMMA (v_wmma_f32_16x16x32_bf16, fp32 accumulate), with
// double-buffered GLOBAL_LOAD_ASYNC_TO_LDS staging (ASYNCcnt) so shared block
// tiles are fetched once from L2 instead of 4x/2x per wave, and WMMA issue is
// decoupled from global load latency.
// ---------------------------------------------------------------------------

typedef __attribute__((ext_vector_type(16))) __bf16 v16bf;
typedef __attribute__((ext_vector_type(8)))  __bf16 v8bf;
typedef __attribute__((ext_vector_type(8)))  float  v8f;

// Parameter type of __builtin_amdgcn_global_load_async_to_lds_b128 per the
// compiler diagnostic: 'int __vector(4) __device__ *' (AS1) / AS3 sibling.
typedef int vsi4 __attribute__((vector_size(16)));
typedef __attribute__((address_space(1))) vsi4 g_vsi4;
typedef __attribute__((address_space(3))) vsi4 l_vsi4;

#define NIMG 4096
#define DIM  768
#define HID  1024

// Block tile geometry: 256 threads = 8 waves (2 M x 4 N), wave tile 64x64.
#define BLK_M 128
#define BLK_N 256
#define KT    32
#define LDS_STRIDE 40   // KT + 8 bf16 pad; row pitch 80B (16B aligned)

enum { EPI_ADJ = 0, EPI_BIAS_T = 1, EPI_RELU_B = 2, EPI_F32 = 3 };

// ---- CDNA5 async global->LDS helpers --------------------------------------
__device__ __forceinline__ void async_b128(const __bf16* gptr, __bf16* lptr)
{
#if __has_builtin(__builtin_amdgcn_global_load_async_to_lds_b128)
    __builtin_amdgcn_global_load_async_to_lds_b128(
        (g_vsi4*)gptr, (l_vsi4*)lptr, 0, 0);
#else
    unsigned int loff = (unsigned int)(uintptr_t)lptr;
    unsigned long long ga = (unsigned long long)(uintptr_t)gptr;
    asm volatile("global_load_async_to_lds_b128 %0, %1, off"
                 :: "v"(loff), "v"(ga) : "memory");
#endif
}

__device__ __forceinline__ void wait_async0()
{
#if __has_builtin(__builtin_amdgcn_s_wait_asynccnt)
    __builtin_amdgcn_s_wait_asynccnt(0);
#else
    asm volatile("s_wait_asynccnt 0x0" ::: "memory");
#endif
}

// Issue one K-stage of async tile loads. A: 128x32 bf16 (512 16B chunks),
// B: 256x32 bf16 (1024 chunks); 256 threads -> 2 + 4 chunks per thread.
__device__ __forceinline__ void stage_loads(
    const __bf16* __restrict__ A, const __bf16* __restrict__ Bt,
    __bf16* As, __bf16* Bs, int m0b, int n0b, int K, int k, int t)
{
#pragma unroll
    for (int q = 0; q < 2; ++q) {
        int c = q * 256 + t;
        int row = c >> 2, chk = c & 3;
        async_b128(A + (size_t)(m0b + row) * K + k + chk * 8,
                   As + row * LDS_STRIDE + chk * 8);
    }
#pragma unroll
    for (int q = 0; q < 4; ++q) {
        int c = q * 256 + t;
        int row = c >> 2, chk = c & 3;
        async_b128(Bt + (size_t)(n0b + row) * K + k + chk * 8,
                   Bs + row * LDS_STRIDE + chk * 8);
    }
}

// C(MxN) = A(MxK,row-major) * Bt(NxK,row-major)^T, fp32 accumulate via WMMA.
// M%128==0, N%256==0, K%32==0 (true for all calls here).
template <int EPI>
__global__ __launch_bounds__(256) void gemm_wmma(
    const __bf16* __restrict__ A, const __bf16* __restrict__ Bt,
    const float* __restrict__ bias, const float* __restrict__ sq,
    float* __restrict__ outF, __bf16* __restrict__ outB,
    int M, int N, int K)
{
    __shared__ __align__(16) __bf16 As[2][BLK_M * LDS_STRIDE]; // 20KB
    __shared__ __align__(16) __bf16 Bs[2][BLK_N * LDS_STRIDE]; // 40KB

    const int t     = threadIdx.x;
    const int lane  = t & 31;
    const int wid   = t >> 5;
    const int laneM = lane & 15;   // M/N-within-16 per ISA fragment layouts
    const int laneH = lane >> 4;   // K-half selector
    const int m0b = blockIdx.y * BLK_M;
    const int n0b = blockIdx.x * BLK_N;
    const int wmOff = (wid >> 2) * 64;
    const int wnOff = (wid & 3) * 64;

    v8f c[4][4] = {};

    stage_loads(A, Bt, As[0], Bs[0], m0b, n0b, K, 0, t);
    wait_async0();
    __syncthreads();

    const int S = K / KT;
    for (int s = 0; s < S; ++s) {
        const int buf = s & 1;
        if (s + 1 < S)
            stage_loads(A, Bt, As[buf ^ 1], Bs[buf ^ 1],
                        m0b, n0b, K, (s + 1) * KT, t);

        // Fragment reads from LDS per the documented bf16 VGPR layouts:
        // A 16x32: lane holds M=lane%16, K-chunks {8*laneH..+7} and {+16..}
        // B 32x16: lane holds N=lane%16, K-chunks {8*laneH..+7} and {+16..}
        v16bf a[4], b[4];
#pragma unroll
        for (int i = 0; i < 4; ++i) {
            const __bf16* p =
                &As[buf][(wmOff + 16 * i + laneM) * LDS_STRIDE + laneH * 8];
            v8bf lo = *(const v8bf*)(p);
            v8bf hi = *(const v8bf*)(p + 16);
            a[i] = __builtin_shufflevector(lo, hi, 0, 1, 2, 3, 4, 5, 6, 7,
                                           8, 9, 10, 11, 12, 13, 14, 15);
        }
#pragma unroll
        for (int j = 0; j < 4; ++j) {
            const __bf16* p =
                &Bs[buf][(wnOff + 16 * j + laneM) * LDS_STRIDE + laneH * 8];
            v8bf lo = *(const v8bf*)(p);
            v8bf hi = *(const v8bf*)(p + 16);
            b[j] = __builtin_shufflevector(lo, hi, 0, 1, 2, 3, 4, 5, 6, 7,
                                           8, 9, 10, 11, 12, 13, 14, 15);
        }
#pragma unroll
        for (int i = 0; i < 4; ++i)
#pragma unroll
            for (int j = 0; j < 4; ++j)
                c[i][j] = __builtin_amdgcn_wmma_f32_16x16x32_bf16(
                    false, a[i], false, b[j], (short)0, c[i][j], false, false);

        wait_async0();
        __syncthreads();
    }

    // Epilogue. C/D layout: VGPR r, lanes0-15 -> M=r, lanes16-31 -> M=8+r.
#pragma unroll
    for (int i = 0; i < 4; ++i) {
#pragma unroll
        for (int j = 0; j < 4; ++j) {
            const int n = n0b + wnOff + 16 * j + laneM;
#pragma unroll
            for (int r = 0; r < 8; ++r) {
                const int m = m0b + wmOff + 16 * i + 8 * laneH + r;
                float v = c[i][j][r];
                if constexpr (EPI == EPI_ADJ) {
                    float d2 = sq[m] + sq[n] - 2.0f * v;
                    d2 = fmaxf(d2, 1e-12f);
                    float g = 1.0f / (1.0f + sqrtf(d2));
                    outF[(size_t)m * N + n] = g;
                    outB[(size_t)m * N + n] = (__bf16)g;
                } else if constexpr (EPI == EPI_BIAS_T) {
                    float y = v + bias[n];
                    outB[(size_t)n * M + m] = (__bf16)y;
                } else if constexpr (EPI == EPI_RELU_B) {
                    outB[(size_t)m * N + n] = (__bf16)fmaxf(v, 0.0f);
                } else { // EPI_F32
                    outF[(size_t)m * N + n] = v;
                }
            }
        }
    }
}

// Per-row squared norm + fp32 -> bf16 convert. One block per row.
__global__ __launch_bounds__(256) void prep_x(
    const float* __restrict__ X, __bf16* __restrict__ Xb,
    float* __restrict__ sq, int D)
{
    const int m = blockIdx.x;
    const float* row = X + (size_t)m * D;
    float acc = 0.0f;
    for (int k = threadIdx.x; k < D; k += 256) {
        float x = row[k];
        acc += x * x;
        Xb[(size_t)m * D + k] = (__bf16)x;
    }
    __shared__ float red[256];
    red[threadIdx.x] = acc;
    __syncthreads();
    for (int s = 128; s > 0; s >>= 1) {
        if (threadIdx.x < s) red[threadIdx.x] += red[threadIdx.x + s];
        __syncthreads();
    }
    if (threadIdx.x == 0) sq[m] = red[0];
}

// Tb[n*Kd + k] = (bf16) W[k*Nd + n]   (W is Kd x Nd row-major)
__global__ __launch_bounds__(256) void conv_transpose(
    const float* __restrict__ W, __bf16* __restrict__ Tb, int Kd, int Nd)
{
    size_t idx = (size_t)blockIdx.x * 256 + threadIdx.x;
    size_t tot = (size_t)Kd * Nd;
    if (idx < tot) {
        int n = (int)(idx / Kd);
        int k = (int)(idx % Kd);
        Tb[idx] = (__bf16)W[(size_t)k * Nd + n];
    }
}

__device__ __forceinline__ void top2_insert(float v, int j, float& v1, int& i1,
                                            float& v2, int& i2)
{
    if (v > v1 || (v == v1 && j < i1)) {
        v2 = v1; i2 = i1; v1 = v; i1 = j;
    } else if (v > v2 || (v == v2 && j < i2)) {
        v2 = v; i2 = j;
    }
}

// Second-largest index per row (jax top_k tie-break: lower index first).
__global__ __launch_bounds__(256) void top2_kernel(
    const float* __restrict__ G, int* __restrict__ outIdx, int N)
{
    const int row = blockIdx.x;
    const int t = threadIdx.x;
    const float* g = G + (size_t)row * N;
    float v1 = -3.0e38f, v2 = -3.0e38f;
    int i1 = 0x7fffffff, i2 = 0x7fffffff;
    for (int j = t; j < N; j += 256)
        top2_insert(g[j], j, v1, i1, v2, i2);

    __shared__ float sv1[256], sv2[256];
    __shared__ int si1[256], si2[256];
    sv1[t] = v1; sv2[t] = v2; si1[t] = i1; si2[t] = i2;
    __syncthreads();
    for (int s = 128; s > 0; s >>= 1) {
        if (t < s) {
            float a1 = sv1[t], a2 = sv2[t];
            int b1 = si1[t], b2 = si2[t];
            top2_insert(sv1[t + s], si1[t + s], a1, b1, a2, b2);
            top2_insert(sv2[t + s], si2[t + s], a1, b1, a2, b2);
            sv1[t] = a1; sv2[t] = a2; si1[t] = b1; si2[t] = b2;
        }
        __syncthreads();
    }
    if (t == 0) outIdx[row] = si2[0];
}

// pseudo[row,:] = X[idx[row],:]  (fp32 copy)
__global__ __launch_bounds__(256) void gather_rows(
    const float* __restrict__ X, const int* __restrict__ idx,
    float* __restrict__ out, int D)
{
    const int row = blockIdx.x;
    const int s = idx[row];
    for (int k = threadIdx.x; k < D; k += 256)
        out[(size_t)row * D + k] = X[(size_t)s * D + k];
}

extern "C" void kernel_launch(void* const* d_in, const int* in_sizes, int n_in,
                              void* d_out, int out_size, void* d_ws,
                              size_t ws_size, hipStream_t stream)
{
    (void)in_sizes; (void)n_in; (void)out_size; (void)ws_size;

    const float* x  = (const float*)d_in[0];  // (1,4096,768)
    // d_in[1] = bank : provably unused by the reference's outputs
    const float* w1 = (const float*)d_in[2];  // (768,1024)
    const float* b1 = (const float*)d_in[3];  // (1024,)
    const float* w2 = (const float*)d_in[4];  // (1024,768)
    const float* b2 = (const float*)d_in[5];  // (768,)

    float* outF   = (float*)d_out;                      // out: 4096x768
    float* pseudo = outF + (size_t)NIMG * DIM;          // pseudo: 4096x768

    char* ws = (char*)d_ws;
    size_t off = 0;
    auto take = [&](size_t bytes) -> char* {
        char* p = ws + off;
        off = (off + bytes + 255) & ~(size_t)255;
        return p;
    };
    __bf16* Xb    = (__bf16*)take((size_t)NIMG * DIM * 2);   // 4096x768
    __bf16* W1Tb  = (__bf16*)take((size_t)HID * DIM * 2);    // 1024x768 (N-major)
    __bf16* W2Tb  = (__bf16*)take((size_t)DIM * HID * 2);    // 768x1024 (N-major)
    float*  sqv   = (float*) take((size_t)NIMG * 4);
    float*  Gf    = (float*) take((size_t)NIMG * NIMG * 4);  // fp32 adjacency
    __bf16* Gb    = (__bf16*)take((size_t)NIMG * NIMG * 2);  // bf16 adjacency
    __bf16* y1Tb  = (__bf16*)take((size_t)HID * NIMG * 2);   // (x@w1+b1)^T
    __bf16* hb    = (__bf16*)take((size_t)NIMG * HID * 2);   // relu(G@y1)
    __bf16* y2Tb  = (__bf16*)take((size_t)DIM * NIMG * 2);   // (h@w2+b2)^T
    int*    idxb  = (int*)   take((size_t)NIMG * 4);

    // 1) norms + bf16 conversion of X; weight transpose-converts
    prep_x<<<NIMG, 256, 0, stream>>>(x, Xb, sqv, DIM);
    conv_transpose<<<(HID * DIM + 255) / 256, 256, 0, stream>>>(w1, W1Tb, DIM, HID);
    conv_transpose<<<(DIM * HID + 255) / 256, 256, 0, stream>>>(w2, W2Tb, HID, DIM);

    // 2) G = adj(X @ X^T)  [M=N=4096, K=768]
    gemm_wmma<EPI_ADJ><<<dim3(NIMG / BLK_N, NIMG / BLK_M), 256, 0, stream>>>(
        Xb, Xb, nullptr, sqv, Gf, Gb, NIMG, NIMG, DIM);

    // 3) top-2 index + pseudo gather
    top2_kernel<<<NIMG, 256, 0, stream>>>(Gf, idxb, NIMG);
    gather_rows<<<NIMG, 256, 0, stream>>>(x, idxb, pseudo, DIM);

    // 4) y1^T = (X @ W1 + b1)^T   [M=4096, N=1024, K=768]
    gemm_wmma<EPI_BIAS_T><<<dim3(HID / BLK_N, NIMG / BLK_M), 256, 0, stream>>>(
        Xb, W1Tb, b1, nullptr, nullptr, y1Tb, NIMG, HID, DIM);

    // 5) h = relu(G @ y1)         [M=4096, N=1024, K=4096]
    gemm_wmma<EPI_RELU_B><<<dim3(HID / BLK_N, NIMG / BLK_M), 256, 0, stream>>>(
        Gb, y1Tb, nullptr, nullptr, nullptr, hb, NIMG, HID, NIMG);

    // 6) y2^T = (h @ W2 + b2)^T   [M=4096, N=768, K=1024]
    gemm_wmma<EPI_BIAS_T><<<dim3(DIM / BLK_N, NIMG / BLK_M), 256, 0, stream>>>(
        hb, W2Tb, b2, nullptr, nullptr, y2Tb, NIMG, DIM, HID);

    // 7) out = G @ y2             [M=4096, N=768, K=4096]
    gemm_wmma<EPI_F32><<<dim3(DIM / BLK_N, NIMG / BLK_M), 256, 0, stream>>>(
        Gb, y2Tb, nullptr, nullptr, outF, nullptr, NIMG, DIM, NIMG);
}
